// SpectralMPOLinear_58402965291431
// MI455X (gfx1250) — compile-verified, matrix operand order checked by reading, not run on hardware
//
#include <hip/hip_runtime.h>

// ---------------------------------------------------------------------------
// SpectralMPOLinear for MI455X (gfx1250, wave32, WMMA).
// y = x @ W_total + bias is materialized as one f16 WMMA GEMM; W_total is
// rebuilt each call from the structured factors (cheap: ~5.7 GMAC fp32 vs
// 137 GFLOP for the GEMM, which is the compute-bound part on this chip).
// ---------------------------------------------------------------------------

typedef __attribute__((ext_vector_type(16))) _Float16 v16h;
typedef __attribute__((ext_vector_type(8)))  _Float16 v8h;
typedef __attribute__((ext_vector_type(4)))  _Float16 h4;
typedef __attribute__((ext_vector_type(8)))  float    v8f;

#define NF    4096          // TOKENS == IN_F == OUT_F
#define HD    128           // head dim (in and out)
#define NNZ   32768
#define TR    64            // tucker rank

// ---------------- 1. DCT matrix  D[k*128+d] = s(k) cos(pi (d+.5) k / 128) ---
__global__ void k_dct(float* __restrict__ D) {
  int idx = blockIdx.x * 256 + threadIdx.x;        // 16384
  int k = idx >> 7, d = idx & 127;
  float s = (k == 0) ? 0.08838834764831845f : 0.125f;  // sqrt(1/128), sqrt(2/128)
  float a = 3.14159265358979323846f * ((float)d + 0.5f) * (float)k * (1.0f / 128.0f);
  D[idx] = s * cosf(a);
}

// ---------------- 2. S1[ij, po] = sum_r mpo_a[i,p,r] * mpo_b[r,j,o] --------
__global__ void k_s1(const float* __restrict__ ma, const float* __restrict__ mb,
                     float* __restrict__ S1) {
  size_t idx = (size_t)blockIdx.x * 256 + threadIdx.x;   // 16.7M
  int row = (int)(idx >> 12);          // ij
  int col = (int)(idx & 4095);         // po
  int i = row >> 6, j = row & 63, p = col >> 6, o = col & 63;
  const float* ap = ma + ((i << 10) + (p << 4));   // mpo_a[i][p][r], (64,64,16)
  const float* bp = mb + ((j << 6) + o);           // mpo_b[r][j][o], (16,64,64)
  float acc = 0.f;
#pragma unroll
  for (int r = 0; r < 16; ++r) acc = fmaf(ap[r], bp[(size_t)r << 12], acc);
  S1[idx] = acc;
}

// ---- 3. right DCT (+ output perm folded as column gather) -----------------
// B[row, c] = sum_e S1[row, head(oip[c])*128 + e] * D[e*128 + (oip[c]&127)]
__global__ void k_right(const float* __restrict__ S1, const float* __restrict__ D,
                        const long long* __restrict__ oip, float* __restrict__ B) {
  __shared__ float rowbuf[NF];
  int row = blockIdx.x;
  const float* src = S1 + (size_t)row * NF;
  for (int t = threadIdx.x; t < NF; t += 256) rowbuf[t] = src[t];
  __syncthreads();
  float* dst = B + (size_t)row * NF;
  for (int c = threadIdx.x; c < NF; c += 256) {
    int g = (int)oip[c];
    int base = g & ~127, gl = g & 127;
    float acc = 0.f;
#pragma unroll 8
    for (int e = 0; e < HD; ++e) acc = fmaf(rowbuf[base + e], D[(e << 7) + gl], acc);
    dst[c] = acc;
  }
}

// ---- 4. left DCT (+ input perm folded as whole-row scatter) ---------------
// Z[input_perm[ha*128+al], c] = sum_k D[k*128+al] * B[ha*128+k, c]
__global__ void k_left(const float* __restrict__ Bm, const float* __restrict__ D,
                       const long long* __restrict__ iperm, float* __restrict__ Z) {
  __shared__ float Bt[HD][32];
  int ha = blockIdx.y;                 // 0..31
  int c0 = blockIdx.x << 5;            // col tile
  int cx = threadIdx.x;                // 0..31
  int ty = threadIdx.y;                // 0..7
  for (int idx = ty * 32 + cx; idx < HD * 32; idx += 256) {
    int k = idx >> 5, cc = idx & 31;
    Bt[k][cc] = Bm[(size_t)(ha * HD + k) * NF + c0 + cc];
  }
  __syncthreads();
  for (int s = 0; s < 16; ++s) {
    int al = (s << 3) + ty;
    float acc = 0.f;
#pragma unroll 8
    for (int k = 0; k < HD; ++k) acc = fmaf(D[(k << 7) + al], Bt[k][cx], acc);
    int arow = (int)iperm[ha * HD + al];
    Z[(size_t)arow * NF + c0 + cx] = acc;
  }
}

// ---- 5. tiled transpose: dst[c,i] = src[i,c] ------------------------------
__global__ void k_transpose(const float* __restrict__ src, float* __restrict__ dst) {
  __shared__ float tile[32][33];
  int x0 = blockIdx.x << 5, y0 = blockIdx.y << 5;
  int tx = threadIdx.x, ty = threadIdx.y;      // 32 x 8
  for (int s = 0; s < 32; s += 8)
    tile[ty + s][tx] = src[(size_t)(y0 + ty + s) * NF + x0 + tx];
  __syncthreads();
  for (int s = 0; s < 32; s += 8)
    dst[(size_t)(x0 + ty + s) * NF + y0 + tx] = tile[tx][ty + s];
}

// ---- 6a. sparse scatter: W[out, in] += v ----------------------------------
__global__ void k_sparse(const long long* __restrict__ sidx,
                         const float* __restrict__ sval, float* __restrict__ W) {
  int t = blockIdx.x * 256 + threadIdx.x;
  if (t < NNZ) {
    long long o = sidx[2 * t], i = sidx[2 * t + 1];
    atomicAdd(&W[(size_t)o * NF + (size_t)i], sval[t]);
  }
}

// ---- 6b. T1[r, i] = sum_r2 core[r, r2] * U_in[i, r2] ----------------------
__global__ void k_t1(const float* __restrict__ core, const float* __restrict__ Uin,
                     float* __restrict__ T1) {
  size_t t = (size_t)blockIdx.x * 256 + threadIdx.x;   // 64*4096
  int r = (int)(t >> 12), i = (int)(t & 4095);
  const float* c = core + r * TR;
  const float* u = Uin + (size_t)i * TR;
  float acc = 0.f;
#pragma unroll
  for (int r2 = 0; r2 < TR; ++r2) acc = fmaf(c[r2], u[r2], acc);
  T1[t] = acc;
}

// ---- 6c. W[c, i] += sum_r U_out[c, r] * T1[r, i] --------------------------
__global__ void k_tuck(const float* __restrict__ Uout, const float* __restrict__ T1,
                       float* __restrict__ W) {
  __shared__ float uo[TR];
  int c = blockIdx.y;
  int i = (blockIdx.x << 8) + threadIdx.x;
  if (threadIdx.x < TR) uo[threadIdx.x] = Uout[(size_t)c * TR + threadIdx.x];
  __syncthreads();
  size_t off = (size_t)c * NF + i;
  float acc = W[off];
#pragma unroll 8
  for (int r = 0; r < TR; ++r) acc = fmaf(uo[r], T1[((size_t)r << 12) + i], acc);
  W[off] = acc;
}

// ---- 7. fp32 -> f16 (vectorized x4) ---------------------------------------
__global__ void k_f2h(const float* __restrict__ s, _Float16* __restrict__ d) {
  size_t t = ((size_t)blockIdx.x * 256 + threadIdx.x) * 4;
  float4 v = *(const float4*)(s + t);
  h4 o;
  o.x = (_Float16)v.x; o.y = (_Float16)v.y; o.z = (_Float16)v.z; o.w = (_Float16)v.w;
  *(h4*)(d + t) = o;
}

// ---- 8. main GEMM: Y[m,n] = sum_k A[m,k]*B[n,k] + bias[n] -----------------
// A = x_h  [4096][4096] (m,k row-major) ; B = W_h [n][k] row-major.
// Block 256 thr (8 waves), tile 128x128, BK=32, wave tile 64x32 (4x2 frags).
// LDS row stride 40 halves: 16B-aligned b128 access, no 64-bank conflicts.
#define BK   32
#define LDP  40
__global__ __launch_bounds__(256) void k_gemm(const _Float16* __restrict__ A,
                                              const _Float16* __restrict__ B,
                                              const float* __restrict__ bias,
                                              float* __restrict__ Y) {
  __shared__ __align__(16) _Float16 As[2][128 * LDP];
  __shared__ __align__(16) _Float16 Bs[2][128 * LDP];

  const int tid = threadIdx.x;
  const int m0 = blockIdx.y << 7;
  const int n0 = blockIdx.x << 7;

  // global->LDS: 2 threads per row, 32B (two b128) each
  const int lr = tid >> 1;
  const int ls = (tid & 1) << 4;

  const int lane = tid & 31;
  const int w    = tid >> 5;
  const int wm   = (w >> 2) << 6;    // 0 / 64
  const int wn   = (w & 3) << 5;     // 0 / 32 / 64 / 96
  const int g    = lane >> 4;        // lane half-group
  const int mr   = lane & 15;
  const int ksel = g << 3;           // ISA 16-bit frag: k = g*8 + e, g*8+16 + e

  v8f c[4][2];
#pragma unroll
  for (int t = 0; t < 4; ++t)
#pragma unroll
    for (int u = 0; u < 2; ++u)
#pragma unroll
      for (int r = 0; r < 8; ++r) c[t][u][r] = 0.f;

  auto load_tile = [&](int buf, int kk) {
    const _Float16* ag = A + (size_t)(m0 + lr) * NF + kk + ls;
    const _Float16* bg = B + (size_t)(n0 + lr) * NF + kk + ls;
    *(v8h*)&As[buf][lr * LDP + ls]     = *(const v8h*)ag;
    *(v8h*)&As[buf][lr * LDP + ls + 8] = *(const v8h*)(ag + 8);
    *(v8h*)&Bs[buf][lr * LDP + ls]     = *(const v8h*)bg;
    *(v8h*)&Bs[buf][lr * LDP + ls + 8] = *(const v8h*)(bg + 8);
  };

  load_tile(0, 0);
  __syncthreads();

  for (int ks = 0; ks < NF / BK; ++ks) {
    const int buf = ks & 1;
    if (ks + 1 < NF / BK) load_tile(buf ^ 1, (ks + 1) * BK);

    v16h a[4], b[2];
#pragma unroll
    for (int t = 0; t < 4; ++t) {
      const _Float16* p = &As[buf][(wm + (t << 4) + mr) * LDP + ksel];
      v8h lo = *(const v8h*)p;
      v8h hi = *(const v8h*)(p + 16);
      a[t] = __builtin_shufflevector(lo, hi, 0,1,2,3,4,5,6,7,8,9,10,11,12,13,14,15);
    }
#pragma unroll
    for (int u = 0; u < 2; ++u) {
      const _Float16* p = &Bs[buf][(wn + (u << 4) + mr) * LDP + ksel];
      v8h lo = *(const v8h*)p;
      v8h hi = *(const v8h*)(p + 16);
      b[u] = __builtin_shufflevector(lo, hi, 0,1,2,3,4,5,6,7,8,9,10,11,12,13,14,15);
    }
#pragma unroll
    for (int t = 0; t < 4; ++t)
#pragma unroll
      for (int u = 0; u < 2; ++u)
        c[t][u] = __builtin_amdgcn_wmma_f32_16x16x32_f16(
            false, a[t], false, b[u], (short)0, c[t][u], false, false);

    __syncthreads();
  }

  // epilogue: C frag layout — VGPR r: M = g*8 + r, N = mr
#pragma unroll
  for (int t = 0; t < 4; ++t)
#pragma unroll
    for (int u = 0; u < 2; ++u) {
      const int row0 = m0 + wm + (t << 4) + (g << 3);
      const int col  = n0 + wn + (u << 4) + mr;
      const float bv = bias[col];
#pragma unroll
      for (int r = 0; r < 8; ++r)
        Y[(size_t)(row0 + r) * NF + col] = c[t][u][r] + bv;
    }
}

// ---------------------------------------------------------------------------
extern "C" void kernel_launch(void* const* d_in, const int* in_sizes, int n_in,
                              void* d_out, int out_size, void* d_ws, size_t ws_size,
                              hipStream_t stream) {
  const float*     x     = (const float*)d_in[0];
  const float*     mpo_a = (const float*)d_in[1];
  const float*     mpo_b = (const float*)d_in[2];
  const float*     core  = (const float*)d_in[3];
  const float*     u_out = (const float*)d_in[4];
  const float*     u_in  = (const float*)d_in[5];
  const float*     sval  = (const float*)d_in[6];
  const float*     bias  = (const float*)d_in[7];
  const long long* sidx  = (const long long*)d_in[8];
  const long long* iperm = (const long long*)d_in[9];
  const long long* operm = (const long long*)d_in[10];
  (void)in_sizes; (void)n_in; (void)out_size; (void)ws_size;

  char* ws = (char*)d_ws;
  float*    bufA = (float*)ws;                                   // 64 MB
  float*    bufB = (float*)(ws + ((size_t)64 << 20));            // 64 MB
  float*    Dm   = (float*)(ws + ((size_t)128 << 20));           // 64 KB
  float*    T1   = (float*)(ws + ((size_t)128 << 20) + (1u << 16)); // 1 MB
  _Float16* Wh   = (_Float16*)bufA;                              // reuse bufA
  _Float16* xh   = (_Float16*)(ws + ((size_t)32 << 20));         // bufA upper half

  k_dct      <<<64, 256, 0, stream>>>(Dm);
  k_s1       <<<65536, 256, 0, stream>>>(mpo_a, mpo_b, bufA);
  k_right    <<<4096, 256, 0, stream>>>(bufA, Dm, operm, bufB);
  k_left     <<<dim3(128, 32), dim3(32, 8), 0, stream>>>(bufB, Dm, iperm, bufA);
  k_transpose<<<dim3(128, 128), dim3(32, 8), 0, stream>>>(bufA, bufB);
  k_sparse   <<<NNZ / 256, 256, 0, stream>>>(sidx, sval, bufB);
  k_t1       <<<(TR * NF) / 256, 256, 0, stream>>>(core, u_in, T1);
  k_tuck     <<<dim3(NF / 256, NF), 256, 0, stream>>>(u_out, T1, bufB);
  k_f2h      <<<(NF * (NF / 4)) / 256, 256, 0, stream>>>(bufB, Wh);
  k_f2h      <<<(NF * (NF / 4)) / 256, 256, 0, stream>>>(x, xh);
  k_gemm     <<<dim3(32, 32), 256, 0, stream>>>(xh, Wh, bias, (float*)d_out);
}
